// MHA_17849884082838
// MI455X (gfx1250) — compile-verified, matrix operand order checked by reading, not run on hardware
//
#include <hip/hip_runtime.h>

// ============================================================================
// MHA forward for MI455X (gfx1250, wave32, WMMA 16x16x32 f16 -> f32 accum).
// B=2, N=2048, D=1024, H=16, HD=64.
//  - GEMMs: v_wmma_f32_16x16x32_f16, 32x64 tile/wave, double-buffered k-loop.
//  - Attention: 4 waves/block share one (b,h); K/V tiles staged once per block
//    into double-buffered LDS (async global->LDS when available, ASYNCcnt
//    drained with s_wait_asynccnt), 4x cut in L2 traffic. Flash-style online
//    softmax, bit-packed mask, P transposed C->A via per-wave LDS slice.
// Workspace layout (assumes ws_size >= 49 MB):
//   [ 0MB) xh     f16 [4096,1024]
//   [ 8MB) wqkvT  f16 [3072,1024]
//   [14MB) woutT  f16 [1024,1024]
//   [16MB) Q      f16 [B,H,N,HD]
//   [24MB) Kd     f16 [B,H,N,HD]
//   [32MB) Vt     f16 [B,H,HD,N]
//   [40MB) AO     f16 [4096,1024]
//   [48MB) maskP  u32 [B,N,N/32]  bit-packed mask
// mask input is a jnp bool array -> treated as 1 byte per element.
// ============================================================================

typedef __attribute__((ext_vector_type(16))) _Float16 v16h;
typedef __attribute__((ext_vector_type(8)))  _Float16 v8h;
typedef __attribute__((ext_vector_type(8)))  float    v8f;
typedef __attribute__((ext_vector_type(4)))  int      v4i;

#define B_   2
#define N_   2048
#define D_   1024
#define H_   16
#define HD_  64
#define NW_  (N_ / 32)         // mask words per row
#define SCALE_ 0.125f          // 64^-0.5
#define NEG_BIG (-1.0e30f)
#define KLD  72                // LDS ld (halves) for K tile rows (64 + 8 pad)
#define VLD  40                // LDS ld (halves) for V tile rows (32 + 8 pad)
#define NSTEP (N_ / 32)        // 64 key steps

#if __has_builtin(__builtin_amdgcn_global_load_async_to_lds_b128)
#define HAS_ASYNC_LDS 1
#else
#define HAS_ASYNC_LDS 0
#endif

// ---------------------------------------------------------------------------
// Fragment loader (ISA 16-bit operand layout):
//   lane l: row = l&15, khalf = 8*(l>>4);
//   halves 0..7  <- K = khalf+0..7, halves 8..15 <- K = 16+khalf+0..7
// Works for A (row=M) and, when the source is B^T, for B (row=N).
// Source may be global or LDS (generic pointer).
// ---------------------------------------------------------------------------
__device__ __forceinline__ v16h load_frag(const _Float16* __restrict__ base, int ld) {
  const int lane = threadIdx.x & 31;
  const int row  = lane & 15;
  const int kh   = (lane >> 4) << 3;   // 0 or 8
  const _Float16* p = base + (size_t)row * ld + kh;
  v8h lo = *(const v8h*)(p);
  v8h hi = *(const v8h*)(p + 16);
  v16h f;
#pragma unroll
  for (int i = 0; i < 8; ++i) { f[i] = lo[i]; f[i + 8] = hi[i]; }
  return f;
}

__device__ __forceinline__ v8f wmma16(v16h a, v16h b, v8f c) {
  return __builtin_amdgcn_wmma_f32_16x16x32_f16(
      false, a, false, b, (short)0, c, false, false);
}

__device__ __forceinline__ float redmax16(float v) {
#pragma unroll
  for (int m = 8; m >= 1; m >>= 1) v = fmaxf(v, __shfl_xor(v, m, 32));
  return v;
}
__device__ __forceinline__ float redsum16(float v) {
#pragma unroll
  for (int m = 8; m >= 1; m >>= 1) v += __shfl_xor(v, m, 32);
  return v;
}

// 16-byte global -> LDS copy: async (ASYNCcnt) when the builtin exists,
// otherwise via VGPRs (global_load_b128 + ds_store_b128).
// Builtin prototype (from hipcc diagnostic): arg0 = v4i addrspace(1)*,
// arg1 = v4i addrspace(3)*, then (imm) offset and cpol.
__device__ __forceinline__ void cp16_g2l(const _Float16* g, _Float16* l) {
#if HAS_ASYNC_LDS
  typedef __attribute__((address_space(1))) v4i* gp_t;
  typedef __attribute__((address_space(3))) v4i* lp_t;
  __builtin_amdgcn_global_load_async_to_lds_b128((gp_t)g, (lp_t)l, 0, 0);
#else
  *(v8h*)l = *(const v8h*)g;
#endif
}

__device__ __forceinline__ void wait_stage(bool next_inflight) {
#if HAS_ASYNC_LDS
  if (next_inflight) asm volatile("s_wait_asynccnt 0x4" ::: "memory");
  else               asm volatile("s_wait_asynccnt 0x0" ::: "memory");
#else
  (void)next_inflight;
#endif
}

// ---------------------------------------------------------------------------
// Prep kernels
// ---------------------------------------------------------------------------
__global__ void cvt_f32_f16(const float* __restrict__ in, _Float16* __restrict__ out, int n) {
  int i = blockIdx.x * blockDim.x + threadIdx.x;
  if (i < n) out[i] = (_Float16)in[i];
}

// in: [R][C] f32 row-major -> out: [C][R] f16 row-major
__global__ void cvt_transpose_f16(const float* __restrict__ in, _Float16* __restrict__ out,
                                  int R, int C) {
  int i = blockIdx.x * blockDim.x + threadIdx.x;
  if (i < R * C) {
    int c = i / R, r = i - c * R;
    out[i] = (_Float16)in[(size_t)r * C + c];
  }
}

// 32 bool bytes -> 1 bitmask word (bools are 0/1, so byte LSB == bool value)
__global__ void pack_mask(const unsigned char* __restrict__ in,
                          unsigned int* __restrict__ out, int nwords) {
  int i = blockIdx.x * blockDim.x + threadIdx.x;
  if (i < nwords) {
    const unsigned long long* p = (const unsigned long long*)(in + (size_t)i * 32);
    unsigned int w = 0;
#pragma unroll
    for (int q = 0; q < 4; ++q) {
      unsigned long long d = p[q];
#pragma unroll
      for (int bbit = 0; bbit < 8; ++bbit)
        w |= (unsigned int)((d >> (8 * bbit)) & 1ull) << (q * 8 + bbit);
    }
    out[i] = w;
  }
}

// ---------------------------------------------------------------------------
// Generic 32x64-tile GEMM body: C[32,64] += A[32,K] * B^T[64,K], f16 in,
// f32 accum, double-buffered k-loop.
// ---------------------------------------------------------------------------
#define GEMM_BODY(A0, A1, Bb, KDIM, ACC)                                      \
  v16h a0 = load_frag((A0), (KDIM));                                          \
  v16h a1 = load_frag((A1), (KDIM));                                          \
  v16h b0 = load_frag((Bb) + (size_t)0  * (KDIM), (KDIM));                    \
  v16h b1 = load_frag((Bb) + (size_t)16 * (KDIM), (KDIM));                    \
  v16h b2 = load_frag((Bb) + (size_t)32 * (KDIM), (KDIM));                    \
  v16h b3 = load_frag((Bb) + (size_t)48 * (KDIM), (KDIM));                    \
  for (int kk = 0; kk < (KDIM); kk += 32) {                                   \
    const int kn = (kk + 32) & ((KDIM) - 1); /* wraps to 0 on last step */    \
    __builtin_prefetch((A0) + (size_t)(threadIdx.x & 15) * (KDIM) + kk + 256, 0, 1); \
    __builtin_prefetch((Bb) + (size_t)(threadIdx.x & 15) * (KDIM) + kk + 256, 0, 1); \
    v16h na0 = load_frag((A0) + kn, (KDIM));                                  \
    v16h na1 = load_frag((A1) + kn, (KDIM));                                  \
    v16h nb0 = load_frag((Bb) + (size_t)0  * (KDIM) + kn, (KDIM));            \
    v16h nb1 = load_frag((Bb) + (size_t)16 * (KDIM) + kn, (KDIM));            \
    v16h nb2 = load_frag((Bb) + (size_t)32 * (KDIM) + kn, (KDIM));            \
    v16h nb3 = load_frag((Bb) + (size_t)48 * (KDIM) + kn, (KDIM));            \
    ACC[0] = wmma16(a0, b0, ACC[0]);                                          \
    ACC[1] = wmma16(a0, b1, ACC[1]);                                          \
    ACC[2] = wmma16(a0, b2, ACC[2]);                                          \
    ACC[3] = wmma16(a0, b3, ACC[3]);                                          \
    ACC[4] = wmma16(a1, b0, ACC[4]);                                          \
    ACC[5] = wmma16(a1, b1, ACC[5]);                                          \
    ACC[6] = wmma16(a1, b2, ACC[6]);                                          \
    ACC[7] = wmma16(a1, b3, ACC[7]);                                          \
    a0 = na0; a1 = na1; b0 = nb0; b1 = nb1; b2 = nb2; b3 = nb3;               \
  }

// ---------------------------------------------------------------------------
// QKV projection: [4096,1024] @ [1024,3072]. 32x64 tile/wave.
// 128 m-tiles x 48 n-tiles = 6144 waves = 768 blocks x 8 waves.
// ---------------------------------------------------------------------------
__global__ __launch_bounds__(256) void qkv_gemm(
    const _Float16* __restrict__ X, const _Float16* __restrict__ Wt,
    _Float16* __restrict__ Qd, _Float16* __restrict__ Kd, _Float16* __restrict__ Vt) {
  const int wid = threadIdx.x >> 5, lane = threadIdx.x & 31;
  const int half = lane >> 4, ln = lane & 15;
  const int tile = blockIdx.x * 8 + wid;
  const int mt = tile & 127;
  const int nt = tile >> 7;
  const _Float16* A0 = X + (size_t)mt * 32 * D_;
  const _Float16* A1 = A0 + 16 * D_;
  const _Float16* Bb = Wt + (size_t)nt * 64 * D_;

  v8f acc[8] = {};
  GEMM_BODY(A0, A1, Bb, D_, acc)

#pragma unroll
  for (int r = 0; r < 2; ++r) {
#pragma unroll
    for (int j = 0; j < 4; ++j) {
#pragma unroll
      for (int v = 0; v < 8; ++v) {
        const int row = mt * 32 + r * 16 + v + half * 8;   // token in [0,4096)
        const int col = nt * 64 + j * 16 + ln;             // qkv col in [0,3072)
        const int b = row >> 11, n = row & (N_ - 1);
        const int sec = col >> 10, c = col & (D_ - 1);
        const int h = c >> 6, hd = c & (HD_ - 1);
        const size_t bh = (size_t)(b * H_ + h);
        const _Float16 val = (_Float16)acc[r * 4 + j][v];
        if (sec == 0)      Qd[(bh * N_ + n) * HD_ + hd] = val;
        else if (sec == 1) Kd[(bh * N_ + n) * HD_ + hd] = val;
        else               Vt[(bh * HD_ + hd) * N_ + n] = val;
      }
    }
  }
}

// ---------------------------------------------------------------------------
// Flash attention. Block = 4 waves sharing one (b,h), each wave a 16-row query
// tile. K/V tiles (32 keys) staged in double-buffered LDS once per block.
// blockIdx.x = bh*32 + qgroup;  qt = qgroup*4 + waveId.
// ---------------------------------------------------------------------------
__global__ __launch_bounds__(128) void attn_kernel(
    const _Float16* __restrict__ Q, const _Float16* __restrict__ K,
    const _Float16* __restrict__ Vt, const unsigned int* __restrict__ maskP,
    _Float16* __restrict__ O) {
  __shared__ __align__(16) _Float16 kbuf[2][32 * KLD];  // keys x hd   (9 KB)
  __shared__ __align__(16) _Float16 vbuf[2][64 * VLD];  // hd x keys  (10 KB)
  __shared__ __align__(16) _Float16 pbuf[4][16 * 40];   // P tiles     (5 KB)

  const int tid = threadIdx.x;
  const int wid = tid >> 5, lane = tid & 31;
  const int half = lane >> 4, ln = lane & 15;
  const int bh = blockIdx.x >> 5;                   // 0..31 (shared by block)
  const int qg = blockIdx.x & 31;
  const int qt = qg * 4 + wid;                      // 0..127
  const int b = bh >> 4, h = bh & (H_ - 1);

  const _Float16* Qb = Q  + ((size_t)bh * N_ + qt * 16) * HD_;
  const _Float16* Kb = K  + (size_t)bh * N_ * HD_;
  const _Float16* Vb = Vt + (size_t)bh * HD_ * N_;
  const unsigned int* Mp = maskP + (size_t)(b * N_ + qt * 16) * NW_;
  _Float16* lds = pbuf[wid];

  // cooperative stage of one 32-key step: 32x64 K tile + 64x32 V tile,
  // 4 x 16B chunks per thread (K: c>>3 = key row; V: c>>2 = hd row)
  auto stage = [&](int buf, int kb) {
    const int c0 = tid, c1 = tid + 128;
    cp16_g2l(Kb + (size_t)(kb + (c0 >> 3)) * HD_ + (c0 & 7) * 8,
             &kbuf[buf][(c0 >> 3) * KLD + (c0 & 7) * 8]);
    cp16_g2l(Kb + (size_t)(kb + (c1 >> 3)) * HD_ + (c1 & 7) * 8,
             &kbuf[buf][(c1 >> 3) * KLD + (c1 & 7) * 8]);
    cp16_g2l(Vb + (size_t)(c0 >> 2) * N_ + kb + (c0 & 3) * 8,
             &vbuf[buf][(c0 >> 2) * VLD + (c0 & 3) * 8]);
    cp16_g2l(Vb + (size_t)(c1 >> 2) * N_ + kb + (c1 & 3) * 8,
             &vbuf[buf][(c1 >> 2) * VLD + (c1 & 3) * 8]);
  };

  const v16h qf0 = load_frag(Qb, HD_);              // hd = 0..31
  const v16h qf1 = load_frag(Qb + 32, HD_);         // hd = 32..63

  v8f acc0 = {}, acc1 = {}, acc2 = {}, acc3 = {};
  float mrow[8], lrow[8];
#pragma unroll
  for (int v = 0; v < 8; ++v) { mrow[v] = NEG_BIG; lrow[v] = 0.0f; }

  stage(0, 0);                                      // prologue stage

  for (int it = 0; it < NSTEP; ++it) {
    const int kb = it * 32;
    const bool more = (it + 1) < NSTEP;
    if (more) stage((it + 1) & 1, kb + 32);         // overlap copy w/ compute
    wait_stage(more);                               // prev stage complete
    __syncthreads();                                // visible to all 4 waves

    const _Float16* kt = kbuf[it & 1];
    const _Float16* vt = vbuf[it & 1];

    // mask words: 1 broadcast u32 per row covers all 32 keys of this step
    unsigned int mw[8];
#pragma unroll
    for (int v = 0; v < 8; ++v)
      mw[v] = Mp[(size_t)(v + half * 8) * NW_ + (kb >> 5)];

    // ---- scores from LDS K tile
    v8f s0 = {}, s1 = {};
    s0 = wmma16(qf0, load_frag(kt, KLD), s0);
    s0 = wmma16(qf1, load_frag(kt + 32, KLD), s0);
    s1 = wmma16(qf0, load_frag(kt + 16 * KLD, KLD), s1);
    s1 = wmma16(qf1, load_frag(kt + 16 * KLD + 32, KLD), s1);

    // ---- scale + mask (C layout: row = v + 8*half, col = ln)
#pragma unroll
    for (int v = 0; v < 8; ++v) {
      s0[v] = ((mw[v] >> ln) & 1u)        ? NEG_BIG : s0[v] * SCALE_;
      s1[v] = ((mw[v] >> (16 + ln)) & 1u) ? NEG_BIG : s1[v] * SCALE_;
    }

    // ---- online softmax update
#pragma unroll
    for (int v = 0; v < 8; ++v) {
      const float rm   = redmax16(fmaxf(s0[v], s1[v]));
      const float mnew = fmaxf(mrow[v], rm);
      const float fac  = __expf(mrow[v] - mnew);
      const float p0   = __expf(s0[v] - mnew);
      const float p1   = __expf(s1[v] - mnew);
      s0[v] = p0; s1[v] = p1;
      lrow[v] = lrow[v] * fac + redsum16(p0 + p1);
      mrow[v] = mnew;
      acc0[v] *= fac; acc1[v] *= fac; acc2[v] *= fac; acc3[v] *= fac;
    }

    // ---- P: C layout -> per-wave LDS slice -> A-operand layout
#pragma unroll
    for (int v = 0; v < 8; ++v) {
      const int r = v + half * 8;
      lds[r * 40 + ln]      = (_Float16)s0[v];
      lds[r * 40 + 16 + ln] = (_Float16)s1[v];
    }
    asm volatile("s_wait_dscnt 0x0" ::: "memory");  // wave-private slice
    const v16h pf = load_frag(lds, 40);

    // ---- PV from LDS V tile
    acc0 = wmma16(pf, load_frag(vt, VLD), acc0);
    acc1 = wmma16(pf, load_frag(vt + 16 * VLD, VLD), acc1);
    acc2 = wmma16(pf, load_frag(vt + 32 * VLD, VLD), acc2);
    acc3 = wmma16(pf, load_frag(vt + 48 * VLD, VLD), acc3);

    __syncthreads();                                // buffer reuse fence
  }

  // ---- normalize + store to [4096, 1024] pre-projection buffer
#pragma unroll
  for (int v = 0; v < 8; ++v) {
    const float inv = 1.0f / lrow[v];
    const size_t row = (size_t)b * N_ + qt * 16 + v + half * 8;
    _Float16* op = O + row * (H_ * HD_) + h * HD_ + ln;
    op[0]  = (_Float16)(acc0[v] * inv);
    op[16] = (_Float16)(acc1[v] * inv);
    op[32] = (_Float16)(acc2[v] * inv);
    op[48] = (_Float16)(acc3[v] * inv);
  }
}

// ---------------------------------------------------------------------------
// Output projection: [4096,1024] @ [1024,1024] + bias -> f32.
// ---------------------------------------------------------------------------
__global__ __launch_bounds__(256) void out_gemm(
    const _Float16* __restrict__ A, const _Float16* __restrict__ Wt,
    const float* __restrict__ bias, float* __restrict__ out) {
  const int wid = threadIdx.x >> 5, lane = threadIdx.x & 31;
  const int half = lane >> 4, ln = lane & 15;
  const int tile = blockIdx.x * 8 + wid;
  const int mt = tile & 127;
  const int nt = tile >> 7;
  const _Float16* A0 = A + (size_t)mt * 32 * D_;
  const _Float16* A1 = A0 + 16 * D_;
  const _Float16* Bb = Wt + (size_t)nt * 64 * D_;

  v8f acc[8] = {};
  GEMM_BODY(A0, A1, Bb, D_, acc)

#pragma unroll
  for (int r = 0; r < 2; ++r) {
#pragma unroll
    for (int j = 0; j < 4; ++j) {
#pragma unroll
      for (int v = 0; v < 8; ++v) {
        const int row = mt * 32 + r * 16 + v + half * 8;
        const int col = nt * 64 + j * 16 + ln;
        out[(size_t)row * D_ + col] = acc[r * 4 + j][v] + bias[col];
      }
    }
  }
}

// ---------------------------------------------------------------------------
extern "C" void kernel_launch(void* const* d_in, const int* in_sizes, int n_in,
                              void* d_out, int out_size, void* d_ws, size_t ws_size,
                              hipStream_t stream) {
  const float*         x     = (const float*)d_in[0];
  const unsigned char* mask  = (const unsigned char*)d_in[1];   // jnp bool -> 1B
  const float*         W_qkv = (const float*)d_in[2];
  const float*         W_out = (const float*)d_in[3];
  const float*         b_out = (const float*)d_in[4];
  float*               out   = (float*)d_out;

  char* ws = (char*)d_ws;
  _Float16*     xh    = (_Float16*)(ws);                          //  8 MB
  _Float16*     wqkvT = (_Float16*)(ws + ((size_t)8  << 20));     //  6 MB
  _Float16*     woutT = (_Float16*)(ws + ((size_t)14 << 20));     //  2 MB
  _Float16*     Qd    = (_Float16*)(ws + ((size_t)16 << 20));     //  8 MB
  _Float16*     Kd    = (_Float16*)(ws + ((size_t)24 << 20));     //  8 MB
  _Float16*     Vt    = (_Float16*)(ws + ((size_t)32 << 20));     //  8 MB
  _Float16*     AO    = (_Float16*)(ws + ((size_t)40 << 20));     //  8 MB
  unsigned int* maskP = (unsigned int*)(ws + ((size_t)48 << 20)); //  1 MB

  const int nx  = B_ * N_ * D_;                // 4194304
  const int nmw = B_ * N_ * NW_;               // 262144 words
  cvt_f32_f16<<<(nx + 255) / 256, 256, 0, stream>>>(x, xh, nx);
  cvt_transpose_f16<<<(D_ * 3 * D_ + 255) / 256, 256, 0, stream>>>(W_qkv, wqkvT, D_, 3 * D_);
  cvt_transpose_f16<<<(D_ * D_ + 255) / 256, 256, 0, stream>>>(W_out, woutT, D_, D_);
  pack_mask<<<(nmw + 255) / 256, 256, 0, stream>>>(mask, maskP, nmw);

  qkv_gemm<<<768, 256, 0, stream>>>(xh, wqkvT, Qd, Kd, Vt);
  attn_kernel<<<1024, 128, 0, stream>>>(Qd, Kd, Vt, maskP, AO);
  out_gemm<<<256, 256, 0, stream>>>(AO, woutT, b_out, out);
}
